// SwinTransformerBlock3D_37194416783491
// MI455X (gfx1250) — matrix operand order repaired and checked by reading, not compiled, not run on hardware
//
#include <hip/hip_runtime.h>
#include <hip/hip_bf16.h>
#include <math.h>

// ---------------------------------------------------------------------------
// Swin Transformer 3D block on gfx1250 (MI455X), f16 WMMA everywhere.
// DIM=256 HEADS=8 WS=4 SS=2 ; B=2 D=16 H=56 W=56
// tokens Ntok = 2*16*56*56 = 100352 ; windows = 1568 ; n=64 ; hd=32
// ---------------------------------------------------------------------------

typedef __attribute__((ext_vector_type(16))) _Float16 v16h;
typedef __attribute__((ext_vector_type(8)))  _Float16 v8h;
typedef __attribute__((ext_vector_type(8)))  float    v8f;

#define NTOK   100352
#define NWIN   1568
#define WINTOK 64
#define CDIM   256
#define NHEAD  8
#define HDIM   32

// CDNA5 async global->LDS (ASYNCcnt) path, if this toolchain declares it.
// Probe round 3: builtin exists; signature wants (AS1 int4*, AS3 int4*, imm, imm).
#if __has_builtin(__builtin_amdgcn_global_load_async_to_lds_b128) && \
    __has_builtin(__builtin_amdgcn_s_wait_asynccnt)
#define USE_ASYNC_LDS 1
typedef int v4i_t __attribute__((vector_size(16)));
typedef __attribute__((address_space(1))) v4i_t gv4i;
typedef __attribute__((address_space(3))) v4i_t lv4i;
#else
#define USE_ASYNC_LDS 0
#endif

// ---------------- WMMA fragment loaders (LDS, row-major, padded ld) --------
// A 16x32 f16: lane m=L%16 ; halves i<8 -> k=(L/16)*8+i ; i>=8 -> k=16+(L/16)*8+(i-8)
__device__ __forceinline__ v16h frag_a(const _Float16* s, int ld, int m0, int k0) {
  int lane = threadIdx.x & 31;
  int m  = m0 + (lane & 15);
  int kb = k0 + ((lane >> 4) << 3);
  v8h lo = *(const v8h*)(s + m * ld + kb);
  v8h hi = *(const v8h*)(s + m * ld + kb + 16);
  return __builtin_shufflevector(lo, hi, 0,1,2,3,4,5,6,7,8,9,10,11,12,13,14,15);
}
// B 32x16 f16, stored transposed in LDS as Bt[n][k]: lane n=L%16 ; i<8 -> k=(L/16)*8+i ; i>=8 -> k=16+(L/16)*8+(i-8)
__device__ __forceinline__ v16h frag_b(const _Float16* s, int ld, int n0, int k0) {
  int lane = threadIdx.x & 31;
  int nn = n0 + (lane & 15);
  int kb = k0 + ((lane >> 4) << 3);
  v8h lo = *(const v8h*)(s + nn * ld + kb);
  v8h hi = *(const v8h*)(s + nn * ld + kb + 16);
  return __builtin_shufflevector(lo, hi, 0,1,2,3,4,5,6,7,8,9,10,11,12,13,14,15);
}

__device__ __forceinline__ v8f wmma_f16(v16h a, v16h b, v8f c) {
  return __builtin_amdgcn_wmma_f32_16x16x32_f16(false, a, false, b, (short)0, c, false, false);
}

// ---------------- weight convert: f32 [K][N] -> f16 transposed [N][K] ------
__global__ void convert_wt_kernel(const float* __restrict__ src,
                                  _Float16* __restrict__ dst, int K, int N) {
  int idx = blockIdx.x * 256 + threadIdx.x;
  if (idx >= K * N) return;
  int k = idx / N, n = idx % N;
  dst[(size_t)n * K + k] = (_Float16)src[idx];
}

// ---------------- LN1 + roll(-2,-2,-2) + window partition -> f16 -----------
// one wave (32 lanes) per token; lane covers 8 channels.
__global__ __launch_bounds__(256) void ln1_window_kernel(
    const float* __restrict__ x, const float* __restrict__ g,
    const float* __restrict__ b, _Float16* __restrict__ xwin) {
  int wv = threadIdx.x >> 5, lane = threadIdx.x & 31;
  int tk = blockIdx.x * 8 + wv;         // window-major token id
  int wi = tk >> 6, t = tk & 63;
  int bb = wi / 784, rr = wi % 784;
  int wd = rr / 196, r3 = rr % 196, wh = r3 / 14, ww = r3 % 14;
  int td = t >> 4, th = (t >> 2) & 3, tw = t & 3;
  int dd = (wd * 4 + td + 2) & 15;      // roll by -ss => source = (shifted+ss)%L
  int hh = (wh * 4 + th + 2) % 56;
  int wo = (ww * 4 + tw + 2) % 56;
  const float* src = x + ((((size_t)bb * 16 + dd) * 56 + hh) * 56 + wo) * CDIM + lane * 8;
  float v[8];
  float4 p0 = *(const float4*)(src);
  float4 p1 = *(const float4*)(src + 4);
  v[0]=p0.x; v[1]=p0.y; v[2]=p0.z; v[3]=p0.w;
  v[4]=p1.x; v[5]=p1.y; v[6]=p1.z; v[7]=p1.w;
  float s = 0.f, s2 = 0.f;
  #pragma unroll
  for (int i = 0; i < 8; i++) { s += v[i]; s2 += v[i] * v[i]; }
  #pragma unroll
  for (int off = 16; off >= 1; off >>= 1) {
    s  += __shfl_xor(s,  off, 32);
    s2 += __shfl_xor(s2, off, 32);
  }
  float mean = s * (1.f / 256.f);
  float var  = s2 * (1.f / 256.f) - mean * mean;
  float inv  = rsqrtf(var + 1e-5f);
  _Float16* dst = xwin + (size_t)tk * CDIM + lane * 8;
  #pragma unroll
  for (int i = 0; i < 8; i++)
    dst[i] = (_Float16)((v[i] - mean) * inv * g[lane * 8 + i] + b[lane * 8 + i]);
}

// ---------------- LN2: x1 f32 -> y f16 (identity token map) ----------------
__global__ __launch_bounds__(256) void ln2_kernel(
    const float* __restrict__ x1, const float* __restrict__ g,
    const float* __restrict__ b, _Float16* __restrict__ y) {
  int wv = threadIdx.x >> 5, lane = threadIdx.x & 31;
  int tk = blockIdx.x * 8 + wv;
  const float* src = x1 + (size_t)tk * CDIM + lane * 8;
  float v[8];
  float4 p0 = *(const float4*)(src);
  float4 p1 = *(const float4*)(src + 4);
  v[0]=p0.x; v[1]=p0.y; v[2]=p0.z; v[3]=p0.w;
  v[4]=p1.x; v[5]=p1.y; v[6]=p1.z; v[7]=p1.w;
  float s = 0.f, s2 = 0.f;
  #pragma unroll
  for (int i = 0; i < 8; i++) { s += v[i]; s2 += v[i] * v[i]; }
  #pragma unroll
  for (int off = 16; off >= 1; off >>= 1) {
    s  += __shfl_xor(s,  off, 32);
    s2 += __shfl_xor(s2, off, 32);
  }
  float mean = s * (1.f / 256.f);
  float var  = s2 * (1.f / 256.f) - mean * mean;
  float inv  = rsqrtf(var + 1e-5f);
  _Float16* dst = y + (size_t)tk * CDIM + lane * 8;
  #pragma unroll
  for (int i = 0; i < 8; i++)
    dst[i] = (_Float16)((v[i] - mean) * inv * g[lane * 8 + i] + b[lane * 8 + i]);
}

// ---------------- tiled WMMA GEMM: C = X(f16 MxK) @ Wt'(f16, [N][K]) -------
// 256 threads = 8 waves ; block tile 128x64 ; K-step 32 ; double-buffered LDS.
// Tile fill uses GLOBAL_LOAD_ASYNC_TO_LDS_B128 (ASYNCcnt) when available so
// the DMA overlaps the WMMAs; falls back to register-staged loads otherwise.
// Wave (wm=w>>1, wn=w&1) owns a 32x32 patch = 2x2 WMMA tiles.
// MODE 0: qkv scatter (+bias, q*=1/sqrt(32))   -> outh (q|k|v blocks)
// MODE 1: proj        (+bias, +x shortcut, window-reverse+roll scatter) -> outf
// MODE 2: fc1         (+bias, exact GELU)      -> outh
// MODE 3: fc2         (+bias, +x1 residual)    -> outf
#define GLDA 40
template <int MODE>
__global__ __launch_bounds__(256) void gemm_kernel(
    const _Float16* __restrict__ X, const _Float16* __restrict__ Wt,
    const float* __restrict__ bias, int K,
    _Float16* __restrict__ outh, float* __restrict__ outf,
    const float* __restrict__ addf) {
  __shared__ _Float16 As[2][128 * GLDA];
  __shared__ _Float16 Bs[2][64 * GLDA];
  int m0 = blockIdx.x * 128, n0 = blockIdx.y * 64;
  int t = threadIdx.x, w = t >> 5, lane = t & 31;
  int wm = w >> 1, wn = w & 1;
  int arow = t >> 1, ah = (t & 1) << 4;       // A: 128 rows x 32 halves
  int brow = t >> 2, bh = (t & 3) << 3;       // B:  64 rows x 32 halves
  const uint4* gA = (const uint4*)(X  + (size_t)(m0 + arow) * K + ah);
  const uint4* gB = (const uint4*)(Wt + (size_t)(n0 + brow) * K + bh);
  int nk = K >> 5;
  v8f acc[2][2] = {};

#if USE_ASYNC_LDS
  // prologue: async tile 0 -> buffer 0
  __builtin_amdgcn_global_load_async_to_lds_b128(
      (gv4i*)(gA + 0), (lv4i*)&As[0][arow * GLDA + ah], 0, 0);
  __builtin_amdgcn_global_load_async_to_lds_b128(
      (gv4i*)(gA + 1), (lv4i*)&As[0][arow * GLDA + ah + 8], 0, 0);
  __builtin_amdgcn_global_load_async_to_lds_b128(
      (gv4i*)(gB + 0), (lv4i*)&Bs[0][brow * GLDA + bh], 0, 0);
  __builtin_amdgcn_s_wait_asynccnt(0);
  __syncthreads();
  for (int it = 0; it < nk; ++it) {
    int buf = it & 1;
    bool more = (it + 1 < nk);
    if (more) {                                 // DMA next tile, overlapped
      int nb = buf ^ 1;
      const uint4* pa = gA + (size_t)(it + 1) * 4;
      const uint4* pb = gB + (size_t)(it + 1) * 4;
      __builtin_amdgcn_global_load_async_to_lds_b128(
          (gv4i*)(pa + 0), (lv4i*)&As[nb][arow * GLDA + ah], 0, 0);
      __builtin_amdgcn_global_load_async_to_lds_b128(
          (gv4i*)(pa + 1), (lv4i*)&As[nb][arow * GLDA + ah + 8], 0, 0);
      __builtin_amdgcn_global_load_async_to_lds_b128(
          (gv4i*)(pb + 0), (lv4i*)&Bs[nb][brow * GLDA + bh], 0, 0);
      if (it + 2 < nk) {                        // L2 prefetch of step+2
        __builtin_prefetch((const void*)(gA + (size_t)(it + 2) * 4), 0, 0);
        __builtin_prefetch((const void*)(gB + (size_t)(it + 2) * 4), 0, 0);
      }
    }
    v16h af0 = frag_a(&As[buf][0], GLDA, (wm * 2 + 0) * 16, 0);
    v16h af1 = frag_a(&As[buf][0], GLDA, (wm * 2 + 1) * 16, 0);
    v16h bf0 = frag_b(&Bs[buf][0], GLDA, (wn * 2 + 0) * 16, 0);
    v16h bf1 = frag_b(&Bs[buf][0], GLDA, (wn * 2 + 1) * 16, 0);
    acc[0][0] = wmma_f16(af0, bf0, acc[0][0]);
    acc[0][1] = wmma_f16(af0, bf1, acc[0][1]);
    acc[1][0] = wmma_f16(af1, bf0, acc[1][0]);
    acc[1][1] = wmma_f16(af1, bf1, acc[1][1]);
    if (more) __builtin_amdgcn_s_wait_asynccnt(0);
    __syncthreads();
  }
#else
  // fallback: register-staged double buffering
  {
    uint4 ra0 = gA[0], ra1 = gA[1];
    uint4 rb0 = gB[0];
    *(uint4*)(&As[0][arow * GLDA + ah])     = ra0;
    *(uint4*)(&As[0][arow * GLDA + ah + 8]) = ra1;
    *(uint4*)(&Bs[0][brow * GLDA + bh])     = rb0;
  }
  __syncthreads();
  for (int it = 0; it < nk; ++it) {
    int buf = it & 1;
    bool more = (it + 1 < nk);
    uint4 ra0, ra1, rb0;
    if (more) {
      const uint4* pa = gA + (size_t)(it + 1) * 4;
      const uint4* pb = gB + (size_t)(it + 1) * 4;
      ra0 = pa[0]; ra1 = pa[1];
      rb0 = pb[0];
      if (it + 2 < nk) {
        __builtin_prefetch((const void*)(gA + (size_t)(it + 2) * 4), 0, 0);
        __builtin_prefetch((const void*)(gB + (size_t)(it + 2) * 4), 0, 0);
      }
    }
    v16h af0 = frag_a(&As[buf][0], GLDA, (wm * 2 + 0) * 16, 0);
    v16h af1 = frag_a(&As[buf][0], GLDA, (wm * 2 + 1) * 16, 0);
    v16h bf0 = frag_b(&Bs[buf][0], GLDA, (wn * 2 + 0) * 16, 0);
    v16h bf1 = frag_b(&Bs[buf][0], GLDA, (wn * 2 + 1) * 16, 0);
    acc[0][0] = wmma_f16(af0, bf0, acc[0][0]);
    acc[0][1] = wmma_f16(af0, bf1, acc[0][1]);
    acc[1][0] = wmma_f16(af1, bf0, acc[1][0]);
    acc[1][1] = wmma_f16(af1, bf1, acc[1][1]);
    if (more) {
      int nb = buf ^ 1;
      *(uint4*)(&As[nb][arow * GLDA + ah])     = ra0;
      *(uint4*)(&As[nb][arow * GLDA + ah + 8]) = ra1;
      *(uint4*)(&Bs[nb][brow * GLDA + bh])     = rb0;
    }
    __syncthreads();
  }
#endif

  #pragma unroll
  for (int bn = 0; bn < 2; bn++) {
    int n = n0 + (wn * 2 + bn) * 16 + (lane & 15);
    float bnv = bias[n];
    #pragma unroll
    for (int a = 0; a < 2; a++) {
      #pragma unroll
      for (int r = 0; r < 8; r++) {
        int m = m0 + (wm * 2 + a) * 16 + r + ((lane >> 4) << 3);
        float val = acc[a][bn][r] + bnv;
        if (MODE == 0) {
          int which = n >> 8, c = n & 255, head = c >> 5, dd = c & 31;
          int wi = m >> 6, tok = m & 63;
          if (which == 0) val *= 0.17677669529663687f;  // hd^-0.5 folded into q
          outh[(size_t)which * ((size_t)NWIN * NHEAD * WINTOK * HDIM) +
               (size_t)wi * (NHEAD * WINTOK * HDIM) + head * (WINTOK * HDIM) +
               tok * HDIM + dd] = (_Float16)val;
        } else if (MODE == 1) {
          int wi = m >> 6, tok = m & 63;
          int bb = wi / 784, rr = wi % 784;
          int wd = rr / 196, r3 = rr % 196, wh = r3 / 14, ww = r3 % 14;
          int td = tok >> 4, th = (tok >> 2) & 3, tw = tok & 3;
          int dd = (wd * 4 + td + 2) & 15;
          int hh = (wh * 4 + th + 2) % 56;
          int wo = (ww * 4 + tw + 2) % 56;
          size_t idx = ((((size_t)bb * 16 + dd) * 56 + hh) * 56 + wo) * CDIM + n;
          outf[idx] = addf[idx] + val;          // shortcut + attention branch
        } else if (MODE == 2) {
          outh[(size_t)m * 1024 + n] =
              (_Float16)(0.5f * val * (1.f + erff(val * 0.70710678118654752f)));
        } else {
          size_t idx = (size_t)m * CDIM + n;
          outf[idx] = addf[idx] + val;          // x1 + MLP branch
        }
      }
    }
  }
}

// ---------------- attention: one block per (window, head) ------------------
__device__ __forceinline__ int reg3(int g, int a, int b) {
  return g < a ? 0 : (g < b ? 1 : 2);
}

__global__ __launch_bounds__(128) void attn_kernel(
    const _Float16* __restrict__ qkv, const float* __restrict__ rpb,
    _Float16* __restrict__ aout) {
  __shared__ _Float16 qs[64 * 40];
  __shared__ _Float16 ks[64 * 40];
  __shared__ _Float16 vt[32 * 72];     // transposed v: vt[d][tok]
  __shared__ float    Sf[64 * 65];
  __shared__ _Float16 Pp[64 * 72];
  __shared__ int      lblS[64];        // shift-mask region label per token
  int wi = blockIdx.x, head = blockIdx.y;
  int t = threadIdx.x, w = t >> 5, lane = t & 31;
  const size_t WQ = (size_t)NWIN * NHEAD * WINTOK * HDIM;
  const _Float16* qg = qkv + (size_t)wi * (NHEAD * WINTOK * HDIM) + head * (WINTOK * HDIM);
  const _Float16* kg = qg + WQ;
  const _Float16* vg = qg + 2 * WQ;
  int lrow = t >> 1, lh = (t & 1) << 4;
  {
    const uint4* g = (const uint4*)(qg + lrow * 32 + lh);
    uint4* l = (uint4*)(qs + lrow * 40 + lh);
    l[0] = g[0]; l[1] = g[1];
  }
  {
    const uint4* g = (const uint4*)(kg + lrow * 32 + lh);
    uint4* l = (uint4*)(ks + lrow * 40 + lh);
    l[0] = g[0]; l[1] = g[1];
  }
  #pragma unroll
  for (int i = 0; i < 16; i++)         // transpose v into LDS
    vt[(lh + i) * 72 + lrow] = vg[lrow * 32 + lh + i];
  int rr = wi % 784;
  int wd = rr / 196, r3 = rr % 196, wh = r3 / 14, ww = r3 % 14;
  if (t < 64) {                        // precompute region labels once
    int td = t >> 4, th = (t >> 2) & 3, tw = t & 3;
    lblS[t] = reg3(wd * 4 + td, 12, 14) * 9 + reg3(wh * 4 + th, 52, 54) * 3 +
              reg3(ww * 4 + tw, 52, 54);
  }
  __syncthreads();

  // S = q @ k^T  (M=64,N=64,K=32): wave owns N strip [w*16, w*16+16)
  v8f acc[4] = {};
  v16h bf = frag_b(ks, 40, w * 16, 0);
  #pragma unroll
  for (int mt = 0; mt < 4; mt++) {
    v16h af = frag_a(qs, 40, mt * 16, 0);
    acc[mt] = wmma_f16(af, bf, acc[mt]);
  }
  // rel-pos bias + shift mask, spill to LDS (padded f32)
  int n = w * 16 + (lane & 15);
  int nd = n >> 4, nh = (n >> 2) & 3, nw = n & 3;
  int lbl_n = lblS[n];
  #pragma unroll
  for (int mt = 0; mt < 4; mt++) {
    #pragma unroll
    for (int r = 0; r < 8; r++) {
      int m = mt * 16 + r + ((lane >> 4) << 3);
      int md = m >> 4, mh = (m >> 2) & 3, mw = m & 3;
      int rpi = ((md - nd + 3) * 7 + (mh - nh + 3)) * 7 + (mw - nw + 3);
      float bias = rpb[rpi * NHEAD + head];
      float maskv = (lblS[m] != lbl_n) ? -100.f : 0.f;
      Sf[m * 65 + n] = acc[mt][r] + bias + maskv;
    }
  }
  __syncthreads();
  // row softmax (64 rows, one thread each)
  if (t < 64) {
    float mx = -1e30f;
    for (int j = 0; j < 64; j++) mx = fmaxf(mx, Sf[t * 65 + j]);
    float sum = 0.f;
    for (int j = 0; j < 64; j++) {
      float e = __expf(Sf[t * 65 + j] - mx);
      Sf[t * 65 + j] = e;
      sum += e;
    }
    float inv = 1.f / sum;
    for (int j = 0; j < 64; j++)
      Pp[t * 72 + j] = (_Float16)(Sf[t * 65 + j] * inv);
  }
  __syncthreads();
  // O = P @ V  (M=64,N=32,K=64): wave owns M tile w
  v8f o[2] = {};
  #pragma unroll
  for (int ks2 = 0; ks2 < 2; ks2++) {
    v16h af = frag_a(Pp, 72, w * 16, ks2 * 32);
    #pragma unroll
    for (int nt = 0; nt < 2; nt++) {
      v16h bf2 = frag_b(vt, 72, nt * 16, ks2 * 32);
      o[nt] = wmma_f16(af, bf2, o[nt]);
    }
  }
  #pragma unroll
  for (int nt = 0; nt < 2; nt++) {
    #pragma unroll
    for (int r = 0; r < 8; r++) {
      int m = w * 16 + r + ((lane >> 4) << 3);
      int d = nt * 16 + (lane & 15);
      aout[((size_t)wi * WINTOK + m) * CDIM + head * HDIM + d] = (_Float16)o[nt][r];
    }
  }
}

// ---------------------------------------------------------------------------
extern "C" void kernel_launch(void* const* d_in, const int* in_sizes, int n_in,
                              void* d_out, int out_size, void* d_ws, size_t ws_size,
                              hipStream_t stream) {
  const float* x       = (const float*)d_in[0];
  const float* n1g     = (const float*)d_in[1];
  const float* n1b     = (const float*)d_in[2];
  const float* qkv_w   = (const float*)d_in[3];
  const float* qkv_b   = (const float*)d_in[4];
  const float* proj_w  = (const float*)d_in[5];
  const float* proj_b  = (const float*)d_in[6];
  const float* rpb     = (const float*)d_in[7];
  const float* n2g     = (const float*)d_in[8];
  const float* n2b     = (const float*)d_in[9];
  const float* fc1_w   = (const float*)d_in[10];
  const float* fc1_b   = (const float*)d_in[11];
  const float* fc2_w   = (const float*)d_in[12];
  const float* fc2_b   = (const float*)d_in[13];
  float* out = (float*)d_out;
  char* ws = (char*)d_ws;

  // workspace layout (bytes)
  _Float16* wt_qkv = (_Float16*)(ws + 0);                         //  768x256 f16
  _Float16* wt_prj = (_Float16*)(ws + 393216);                    //  256x256 f16
  _Float16* wt_fc1 = (_Float16*)(ws + 524288);                    // 1024x256 f16
  _Float16* wt_fc2 = (_Float16*)(ws + 1048576);                   //  256x1024 f16
  _Float16* xwin   = (_Float16*)(ws + 1572864);                   // 100352x256 f16 (later: y)
  _Float16* qkvb   = (_Float16*)(ws + 52953088);                  // 3x q/k/v  f16 (later: h low)
  _Float16* attno  = (_Float16*)(ws + 207093760);                 // 100352x256 f16 (later: h high)
  _Float16* hbuf   = (_Float16*)(ws + 52953088);                  // 100352x1024 f16 (reuse qkv+attno)
  float*    x1     = (float*)   (ws + 258473984);                 // 100352x256 f32

  // 1) weights -> f16 transposed [N][K]
  convert_wt_kernel<<<(256 * 768  + 255) / 256, 256, 0, stream>>>(qkv_w,  wt_qkv, 256, 768);
  convert_wt_kernel<<<(256 * 256  + 255) / 256, 256, 0, stream>>>(proj_w, wt_prj, 256, 256);
  convert_wt_kernel<<<(256 * 1024 + 255) / 256, 256, 0, stream>>>(fc1_w,  wt_fc1, 256, 1024);
  convert_wt_kernel<<<(1024 * 256 + 255) / 256, 256, 0, stream>>>(fc2_w,  wt_fc2, 1024, 256);

  // 2) LN1 + roll + window partition
  ln1_window_kernel<<<NTOK / 8, 256, 0, stream>>>(x, n1g, n1b, xwin);

  // 3) QKV GEMM: [100352,256] x [256,768]
  gemm_kernel<0><<<dim3(NTOK / 128, 768 / 64), 256, 0, stream>>>(
      xwin, wt_qkv, qkv_b, 256, qkvb, nullptr, nullptr);

  // 4) windowed attention (1568 windows x 8 heads)
  attn_kernel<<<dim3(NWIN, NHEAD), 128, 0, stream>>>(qkvb, rpb, attno);

  // 5) proj GEMM + window-reverse + roll + shortcut -> x1
  gemm_kernel<1><<<dim3(NTOK / 128, 256 / 64), 256, 0, stream>>>(
      attno, wt_prj, proj_b, 256, nullptr, x1, x);

  // 6) LN2 -> y (reuses xwin region)
  ln2_kernel<<<NTOK / 8, 256, 0, stream>>>(x1, n2g, n2b, xwin);

  // 7) fc1 GEMM + GELU -> h
  gemm_kernel<2><<<dim3(NTOK / 128, 1024 / 64), 256, 0, stream>>>(
      xwin, wt_fc1, fc1_b, 256, hbuf, nullptr, nullptr);

  // 8) fc2 GEMM + x1 residual -> out
  gemm_kernel<3><<<dim3(NTOK / 128, 256 / 64), 256, 0, stream>>>(
      hbuf, wt_fc2, fc2_b, 1024, nullptr, out, x1);
}
